// MambaBlock_8607114461293
// MI455X (gfx1250) — matrix operand order repaired and checked by reading, not compiled, not run on hardware
//
#include <hip/hip_runtime.h>
#include <hip/hip_bf16.h>
#include <math.h>

// ---------------------------------------------------------------------------
// Mamba block for MI455X (gfx1250), wave32, WMMA bf16 path + async-tensor LDS.
//   B=4, L=2048, D_MODEL=1024, D_INNER=2048, D_STATE=16, DT_RANK=64, D_CONV=4
// Pipeline:
//   x,W -> bf16 ; xz = x @ in_w^T (WMMA, bf16 out)
//   depthwise causal conv(k=4)+bias -> xconv (bf16)
//   x_dbl = xconv @ xproj_w_pad^T (WMMA, fp32 out, N padded 96->128)
//   dt_mm = x_dbl[:, :64] @ dt_w^T (WMMA, fp32 out)
//   selective scan (fp32 state, exp2-based dA), fused u*D + silu(z) gate -> y
//   out = y @ out_w^T (WMMA, fp32 out -> d_out)
// GEMM staging uses GLOBAL_LOAD_ASYNC_TO_LDS_B128 (ASYNCcnt) with double-
// buffered LDS: one barrier + one s_wait_asynccnt per k-step.
// ---------------------------------------------------------------------------

typedef __attribute__((ext_vector_type(16))) __bf16 v16bf;
typedef __attribute__((ext_vector_type(8)))  __bf16 v8bf;
typedef __attribute__((ext_vector_type(8)))  float  v8f;

#define LDSTRIDE 40           // 32 halfs + 8 pad (80B row stride, 16B chunks)
#define BUFELEMS (192 * LDSTRIDE)   // A(128 rows) + B(64 rows) per buffer

// async global -> LDS, 16 bytes per lane (VDST = LDS byte address VGPR)
__device__ inline void async_cp16(unsigned lds_byte_off, const void* gaddr) {
  asm volatile("global_load_async_to_lds_b128 %0, %1, off"
               :: "v"(lds_byte_off),
                  "v"((unsigned long long)(uintptr_t)gaddr)
               : "memory");
}
__device__ inline void wait_async0() {
  asm volatile("s_wait_asynccnt 0x0" ::: "memory");
}

// ----------------------- WMMA fragment loaders (LDS) -----------------------
// A (16x32 bf16, MxK): lane m=lane&15; lanes<16 hold K={0..7,16..23},
// lanes>=16 hold K={8..15,24..31}  (ISA 7.12.2 16-bit A layout)
__device__ inline v16bf frag_a(const __bf16* t, int lane) {
  int m = lane & 15, ko = (lane >> 4) * 8;
  const __bf16* p = t + m * LDSTRIDE + ko;
  v8bf lo = *(const v8bf*)(p);
  v8bf hi = *(const v8bf*)(p + 16);
  return __builtin_shufflevector(lo, hi, 0,1,2,3,4,5,6,7,8,9,10,11,12,13,14,15);
}
// B (32x16 bf16, KxN): lane n=lane&15; lanes<16 hold K={0..7,16..23},
// lanes>=16 hold K={8..15,24..31}
__device__ inline v16bf frag_b(const __bf16* t, int lane) {
  int n = lane & 15, ko = (lane >> 4) * 8;
  const __bf16* p = t + n * LDSTRIDE + ko;
  v8bf lo = *(const v8bf*)(p);
  v8bf hi = *(const v8bf*)(p + 16);
  return __builtin_shufflevector(lo, hi, 0,1,2,3,4,5,6,7,8,9,10,11,12,13,14,15);
}

// -------------------------- tiled bf16 WMMA GEMM ---------------------------
// C[M,N] = A[M,K] * Bw[N,K]^T ; block tile 128x64, K-step 32, 8 waves (4x2),
// each wave 32x32 = 2x2 v_wmma_f32_16x16x32_bf16. Requires M%128==0, N%64==0,
// K%32==0. Double-buffered async LDS staging.
template<bool OUT_BF16>
__global__ __launch_bounds__(256)
void gemm_bf16(const __bf16* __restrict__ A, const __bf16* __restrict__ Bw,
               void* __restrict__ Cout, int M, int N, int K)
{
  // single shared object -> LDS offset 0, so element offsets == LDS offsets/2
  __shared__ __align__(16) __bf16 smem[2 * BUFELEMS];
  const int tid  = threadIdx.x;
  const int lane = tid & 31, wave = tid >> 5;
  const int waveM = wave >> 1, waveN = wave & 1;
  const int rowBase = blockIdx.y * 128;
  const int colBase = blockIdx.x * 64;

  v8f acc00{}, acc01{}, acc10{}, acc11{};

  // stage one 128x32 A tile + 64x32 B tile into buffer at element off bufOff
  auto stage = [&](int bufOff, int kk) {
    int q = tid;                              // A: 128 rows * 4 quads = 512
    #pragma unroll
    for (int it = 0; it < 2; ++it, q += 256) {
      int r = q >> 2, c = (q & 3) * 8;
      async_cp16(2u * (unsigned)(bufOff + r * LDSTRIDE + c),
                 &A[(size_t)(rowBase + r) * K + kk + c]);
    }
    int r = tid >> 2, c = (tid & 3) * 8;      // B: 64 rows * 4 quads = 256
    async_cp16(2u * (unsigned)(bufOff + 128 * LDSTRIDE + r * LDSTRIDE + c),
               &Bw[(size_t)(colBase + r) * K + kk + c]);
  };

  stage(0, 0);
  wait_async0();
  __syncthreads();

  const int nk = K / 32;
  for (int i = 0; i < nk; ++i) {
    const int cur = (i & 1) * BUFELEMS;
    const int nxt = ((i + 1) & 1) * BUFELEMS;
    if (i + 1 < nk) stage(nxt, (i + 1) * 32);   // overlap with WMMA below

    const __bf16* Ab = smem + cur + (waveM * 32) * LDSTRIDE;
    const __bf16* Bb = smem + cur + 128 * LDSTRIDE + (waveN * 32) * LDSTRIDE;
    v16bf a0 = frag_a(Ab, lane);
    v16bf a1 = frag_a(Ab + 16 * LDSTRIDE, lane);
    v16bf b0 = frag_b(Bb, lane);
    v16bf b1 = frag_b(Bb + 16 * LDSTRIDE, lane);
    acc00 = __builtin_amdgcn_wmma_f32_16x16x32_bf16(false, a0, false, b0, (short)0, acc00, false, false);
    acc01 = __builtin_amdgcn_wmma_f32_16x16x32_bf16(false, a0, false, b1, (short)0, acc01, false, false);
    acc10 = __builtin_amdgcn_wmma_f32_16x16x32_bf16(false, a1, false, b0, (short)0, acc10, false, false);
    acc11 = __builtin_amdgcn_wmma_f32_16x16x32_bf16(false, a1, false, b1, (short)0, acc11, false, false);

    wait_async0();        // next tile landed in LDS
    __syncthreads();      // and everyone is done reading the current tile
  }

  // epilogue: C/D layout — VGPR r holds row r (lanes 0-15) / row 8+r (16-31)
  int cn = lane & 15, rb = (lane >> 4) * 8;
  int r0 = rowBase + waveM * 32, c0 = colBase + waveN * 32;
  #pragma unroll
  for (int r = 0; r < 8; ++r) {
    int row0 = r0 + rb + r;
    int row1 = r0 + 16 + rb + r;
    if (OUT_BF16) {
      __bf16* C = (__bf16*)Cout;
      C[(size_t)row0 * N + c0 + cn]      = (__bf16)acc00[r];
      C[(size_t)row0 * N + c0 + 16 + cn] = (__bf16)acc01[r];
      C[(size_t)row1 * N + c0 + cn]      = (__bf16)acc10[r];
      C[(size_t)row1 * N + c0 + 16 + cn] = (__bf16)acc11[r];
    } else {
      float* C = (float*)Cout;
      C[(size_t)row0 * N + c0 + cn]      = acc00[r];
      C[(size_t)row0 * N + c0 + 16 + cn] = acc01[r];
      C[(size_t)row1 * N + c0 + cn]      = acc10[r];
      C[(size_t)row1 * N + c0 + 16 + cn] = acc11[r];
    }
  }
}

// ------------------------------ prep kernels -------------------------------
__global__ void k_f32_to_bf16(const float* __restrict__ in, __bf16* __restrict__ out, int n) {
  int i = blockIdx.x * 256 + threadIdx.x;
  if (i < n) out[i] = (__bf16)in[i];
}

__global__ void k_pad_xprojw(const float* __restrict__ w, __bf16* __restrict__ out) {
  int i = blockIdx.x * 256 + threadIdx.x;        // 128*2048
  int r = i >> 11, c = i & 2047;
  out[i] = (r < 96) ? (__bf16)w[r * 2048 + c] : (__bf16)0.0f;
}

__global__ void k_extract_dt(const float* __restrict__ xdbl, __bf16* __restrict__ out) {
  int i = blockIdx.x * 256 + threadIdx.x;        // 8192*64
  int m = i >> 6, c = i & 63;
  out[i] = (__bf16)xdbl[(size_t)m * 128 + c];
}

__global__ void k_prep_A2(const float* __restrict__ A_log, float* __restrict__ A2) {
  int i = blockIdx.x * 256 + threadIdx.x;        // 2048*16
  if (i < 2048 * 16) A2[i] = -__expf(A_log[i]) * 1.4426950408889634f; // A*log2(e)
}

// depthwise causal conv (k=4) + bias; xx = xz[:, :2048]
__global__ __launch_bounds__(256)
void k_conv(const __bf16* __restrict__ xz, const float* __restrict__ cw,
            const float* __restrict__ cb, __bf16* __restrict__ xconv) {
  int i = blockIdx.x * 256 + threadIdx.x;        // 8192*2048
  int d = i & 2047, m = i >> 11;
  int l = m & 2047, b = m >> 11;
  float acc = cb[d];
  #pragma unroll
  for (int j = 0; j < 4; ++j) {
    int l2 = l + j - 3;
    if (l2 >= 0)
      acc += cw[d * 4 + j] * (float)xz[((size_t)(b * 2048 + l2)) * 4096 + d];
  }
  xconv[i] = (__bf16)acc;
}

// ------------------------------ selective scan -----------------------------
// One thread per (b,d); h[16] in registers; B/C staged in LDS per 64 steps.
// delta = softplus(mm + 2*dt_b) (matches reference's double bias),
// dA = exp2(delta * A*log2e), fused u*D + silu(z) gate, bf16 output.
__global__ __launch_bounds__(256)
void k_scan(const float* __restrict__ mm, const __bf16* __restrict__ xconv,
            const __bf16* __restrict__ xz, const float* __restrict__ xdbl,
            const float* __restrict__ A2, const float* __restrict__ dtb,
            const float* __restrict__ Dv, __bf16* __restrict__ ybf)
{
  __shared__ float BC[64][32];                   // [step][B:0..15 | C:16..31]
  const int d = blockIdx.x * 256 + threadIdx.x;  // 0..2047
  const int b = blockIdx.y;                      // 0..3
  float a2[16], h[16];
  #pragma unroll
  for (int n = 0; n < 16; ++n) { a2[n] = A2[d * 16 + n]; h[n] = 0.f; }
  const float bias2 = 2.f * dtb[d];
  const float Dd = Dv[d];

  for (int c = 0; c < 32; ++c) {
    __syncthreads();
    for (int i = threadIdx.x; i < 64 * 32; i += 256) {
      int s = i >> 5, n = i & 31;                // cols 64..95 of x_dbl = B|C
      BC[s][n] = xdbl[((size_t)(b * 2048 + c * 64 + s)) * 128 + 64 + n];
    }
    __syncthreads();
    for (int s = 0; s < 64; ++s) {
      size_t row = (size_t)(b * 2048 + c * 64 + s);
      float dpre  = mm[row * 2048 + d] + bias2;
      float delta = (dpre > 20.f) ? dpre : log1pf(__expf(dpre));
      float xc = (float)xconv[row * 2048 + d];
      float u  = xc / (1.f + __expf(-xc));       // silu(xconv)
      float du = delta * u;
      float y = 0.f;
      #pragma unroll
      for (int n = 0; n < 16; ++n) {
        float dA = __builtin_amdgcn_exp2f(delta * a2[n]);
        h[n] = h[n] * dA + du * BC[s][n];
        y += h[n] * BC[s][16 + n];
      }
      y += u * Dd;
      float z = (float)xz[row * 4096 + 2048 + d];
      float g = z / (1.f + __expf(-z));          // silu(z)
      ybf[row * 2048 + d] = (__bf16)(y * g);
    }
  }
}

// ------------------------------ host launcher ------------------------------
extern "C" void kernel_launch(void* const* d_in, const int* in_sizes, int n_in,
                              void* d_out, int out_size, void* d_ws, size_t ws_size,
                              hipStream_t stream) {
  const float* x       = (const float*)d_in[0]; // [8192,1024]
  const float* in_w    = (const float*)d_in[1]; // [4096,1024]
  const float* conv_w  = (const float*)d_in[2]; // [2048,1,4]
  const float* conv_b  = (const float*)d_in[3]; // [2048]
  const float* xproj_w = (const float*)d_in[4]; // [96,2048]
  const float* dt_w    = (const float*)d_in[5]; // [2048,64]
  const float* dt_b    = (const float*)d_in[6]; // [2048]
  const float* A_log   = (const float*)d_in[7]; // [2048,16]
  const float* Dvec    = (const float*)d_in[8]; // [2048]
  const float* out_w   = (const float*)d_in[9]; // [1024,2048]

  char* ws = (char*)d_ws;
  size_t off = 0;
  auto take = [&](size_t bytes) {
    char* p = ws + off;
    off += (bytes + 255) & ~(size_t)255;
    return p;
  };
  __bf16* x_bf     = (__bf16*)take((size_t)8192 * 1024 * 2);
  __bf16* inw_bf   = (__bf16*)take((size_t)4096 * 1024 * 2);
  __bf16* xz_bf    = (__bf16*)take((size_t)8192 * 4096 * 2);
  __bf16* xconv_bf = (__bf16*)take((size_t)8192 * 2048 * 2);
  __bf16* xpw_bf   = (__bf16*)take((size_t)128 * 2048 * 2);
  float*  xdbl     = (float*) take((size_t)8192 * 128 * 4);
  __bf16* dt_bf    = (__bf16*)take((size_t)8192 * 64 * 2);
  __bf16* dtw_bf   = (__bf16*)take((size_t)2048 * 64 * 2);
  float*  mm       = (float*) take((size_t)8192 * 2048 * 4);
  float*  A2       = (float*) take((size_t)2048 * 16 * 4);
  __bf16* y_bf     = (__bf16*)take((size_t)8192 * 2048 * 2);
  __bf16* outw_bf  = (__bf16*)take((size_t)1024 * 2048 * 2);

  // 1) precision conversion
  k_f32_to_bf16<<<(8192 * 1024) / 256, 256, 0, stream>>>(x, x_bf, 8192 * 1024);
  k_f32_to_bf16<<<(4096 * 1024) / 256, 256, 0, stream>>>(in_w, inw_bf, 4096 * 1024);
  k_f32_to_bf16<<<(2048 * 64) / 256, 256, 0, stream>>>(dt_w, dtw_bf, 2048 * 64);
  k_f32_to_bf16<<<(1024 * 2048) / 256, 256, 0, stream>>>(out_w, outw_bf, 1024 * 2048);
  k_pad_xprojw<<<(128 * 2048) / 256, 256, 0, stream>>>(xproj_w, xpw_bf);
  k_prep_A2<<<(2048 * 16) / 256, 256, 0, stream>>>(A_log, A2);

  // 2) in_proj: xz[8192,4096] (bf16 out)
  gemm_bf16<true><<<dim3(4096 / 64, 8192 / 128), 256, 0, stream>>>(
      x_bf, inw_bf, (void*)xz_bf, 8192, 4096, 1024);

  // 3) depthwise causal conv + bias
  k_conv<<<(8192 * 2048) / 256, 256, 0, stream>>>(xz_bf, conv_w, conv_b, xconv_bf);

  // 4) x_proj: x_dbl[8192,128pad] (fp32 out)
  gemm_bf16<false><<<dim3(128 / 64, 8192 / 128), 256, 0, stream>>>(
      xconv_bf, xpw_bf, (void*)xdbl, 8192, 128, 2048);

  // 5) dt_proj: mm[8192,2048] (fp32 out; bias folded into scan)
  k_extract_dt<<<(8192 * 64) / 256, 256, 0, stream>>>(xdbl, dt_bf);
  gemm_bf16<false><<<dim3(2048 / 64, 8192 / 128), 256, 0, stream>>>(
      dt_bf, dtw_bf, (void*)mm, 8192, 2048, 64);

  // 6) selective scan + D skip + silu(z) gate -> y (bf16)
  k_scan<<<dim3(2048 / 256, 4), 256, 0, stream>>>(
      mm, xconv_bf, xz_bf, xdbl, A2, dt_b, Dvec, y_bf);

  // 7) out_proj -> d_out (fp32)
  gemm_bf16<false><<<dim3(1024 / 64, 8192 / 128), 256, 0, stream>>>(
      y_bf, outw_bf, d_out, 8192, 1024, 2048);
}
